// MyModelAll_34376918238011
// MI455X (gfx1250) — matrix operand order repaired and checked by reading, not compiled, not run on hardware
//
#include <hip/hip_runtime.h>
#include <hip/hip_bf16.h>
#include <math.h>

// ---------------------------------------------------------------------------
// Problem constants: B=256 images = 256 captions, R=36 regions, L=40 words,
// D=1024 embed. Output 256x256 fp32.
// ---------------------------------------------------------------------------
#define NB   256
#define RR   36
#define LLW  40
#define DD   1024
#define TI   2            // images per workgroup
#define TC   2            // captions per workgroup
#define KC   64           // K-chunk staged through LDS (halfs)
#define KS   (KC + 8)     // half stride: 72 halfs = 144B (16B multiple, bank-spread)
#define SS   52           // fp32 S-tile stride
#define NTH  320          // 10 wave32; waves 0..8 do the 3x3 WMMA tiling

typedef __attribute__((ext_vector_type(16))) _Float16 v16h;
typedef __attribute__((ext_vector_type(8)))  _Float16 v8h;
typedef __attribute__((ext_vector_type(8)))  float    v8f;
typedef int v4i_b __attribute__((vector_size(16)));   // matches builtin proto

#if __has_builtin(__builtin_amdgcn_global_load_async_to_lds_b128)
#define ASYNC_CP 1
#define ASYNC_CP_B128(SRC, DST)                                               \
    __builtin_amdgcn_global_load_async_to_lds_b128(                           \
        (__attribute__((address_space(1))) v4i_b*)(SRC),                      \
        (__attribute__((address_space(3))) v4i_b*)(DST), 0, 0)
#endif

// ---------------------------------------------------------------------------
// fp32 -> f16 one-shot conversion (so the pair kernel streams f16 and can use
// async 16B global->LDS copies).
// ---------------------------------------------------------------------------
__global__ __launch_bounds__(256) void cvt_f16_kernel(
    const float* __restrict__ in, _Float16* __restrict__ out, int n)
{
    for (int i = blockIdx.x * blockDim.x + threadIdx.x; i < n;
         i += gridDim.x * blockDim.x)
        out[i] = (_Float16)in[i];
}

// ---------------------------------------------------------------------------
// out[row * 256 + tid] = dot(rows[row], vecs[tid]) over D=1024.
// Used for base, P = img_emb . pool_txt, Q = cap_emb . pool_img.
// ---------------------------------------------------------------------------
__global__ __launch_bounds__(256) void dot_rv_kernel(
    const float* __restrict__ rows, const float* __restrict__ vecs,
    float* __restrict__ out)
{
    const float4* a = (const float4*)(rows + (size_t)blockIdx.x * DD);
    const float4* v = (const float4*)(vecs + (size_t)threadIdx.x * DD);
    float s = 0.0f;
    #pragma unroll 4
    for (int k = 0; k < DD / 4; ++k) {
        float4 x = a[k], y = v[k];
        s = fmaf(x.x, y.x, s); s = fmaf(x.y, y.y, s);
        s = fmaf(x.z, y.z, s); s = fmaf(x.w, y.w, s);
    }
    out[(size_t)blockIdx.x * blockDim.x + threadIdx.x] = s;
}

// ---------------------------------------------------------------------------
// Per-batch n x n Gram matrices: out[b][r][r2] = x[b,r,:] . x[b,r2,:]
// ---------------------------------------------------------------------------
__global__ __launch_bounds__(256) void gram_kernel(
    const float* __restrict__ x, float* __restrict__ out, int n)
{
    const float* base = x + (size_t)blockIdx.x * n * DD;
    for (int p = threadIdx.x; p < n * n; p += blockDim.x) {
        int r = p / n, r2 = p % n;
        const float4* a = (const float4*)(base + (size_t)r  * DD);
        const float4* c = (const float4*)(base + (size_t)r2 * DD);
        float s = 0.0f;
        #pragma unroll 4
        for (int k = 0; k < DD / 4; ++k) {
            float4 xx = a[k], yy = c[k];
            s = fmaf(xx.x, yy.x, s); s = fmaf(xx.y, yy.y, s);
            s = fmaf(xx.z, yy.z, s); s = fmaf(xx.w, yy.w, s);
        }
        out[(size_t)blockIdx.x * n * n + p] = s;
    }
}

// ---------------------------------------------------------------------------
// Pooled-vector L2 norms. Launch <<<2,256>>>.
// ---------------------------------------------------------------------------
__global__ __launch_bounds__(256) void norm_kernel(
    const float* __restrict__ pimg, const float* __restrict__ ptxt,
    float* __restrict__ w1img, float* __restrict__ w1txt)
{
    int t = blockIdx.x * blockDim.x + threadIdx.x;
    const float4* src = (t < NB) ? (const float4*)(pimg + (size_t)t * DD)
                                 : (const float4*)(ptxt + (size_t)(t - NB) * DD);
    float s = 0.0f;
    #pragma unroll 4
    for (int k = 0; k < DD / 4; ++k) {
        float4 x = src[k];
        s = fmaf(x.x, x.x, s); s = fmaf(x.y, x.y, s);
        s = fmaf(x.z, x.z, s); s = fmaf(x.w, x.w, s);
    }
    s = sqrtf(s);
    if (t < NB) w1img[t] = s; else w1txt[t - NB] = s;
}

// ---------------------------------------------------------------------------
// Fused pair kernel: one 320-thread WG per (2 images x 2 captions) group.
// Waves 0..8 tile the padded 48x48 score matrices as 3x3 WMMA 16x16 tiles;
// each wave holds 4 accumulators (2x2 pairs), so every A/B fragment feeds two
// v_wmma_f32_16x16x32_f16 ops. f16 operands staged global->LDS (async copy
// when available). Postprocessing: both SCAN attention paths via P/Q dot
// tables and G/H Gram quadratic forms (no explicit context vectors).
// ---------------------------------------------------------------------------
__global__ __launch_bounds__(NTH) void pair_kernel(
    const _Float16* __restrict__ imgH, // (256,36,1024) f16
    const _Float16* __restrict__ capH, // (256,40,1024) f16
    const float* __restrict__ P,       // (256,36,256)
    const float* __restrict__ Q,       // (256,40,256)
    const float* __restrict__ G,       // (256,36,36)
    const float* __restrict__ H,       // (256,40,40)
    const float* __restrict__ basem,   // (256,256)
    const float* __restrict__ w1img,   // (256,)
    const float* __restrict__ w1txt,   // (256,)
    float* __restrict__ out)           // (256,256)
{
    __shared__ _Float16 Ah[TI * 48 * KS];       // 13.5 KB
    __shared__ _Float16 Bh[TC * 48 * KS];       // 13.5 KB
    __shared__ float    Sl[TI * TC * 48 * SS];  // 39 KB (4 score tiles)
    __shared__ float    Gs[TI * RR * RR];       // 10.1 KB
    __shared__ float    Hs[TC * LLW * LLW];     // 12.5 KB
    __shared__ float    invRow[TI * TC * RR];
    __shared__ float    invCol[TI * TC * LLW];
    __shared__ float    redT[TI * TC * LLW];
    __shared__ float    redI[TI * TC * RR];

    const int tid = threadIdx.x;
    const int b0  = (blockIdx.x >> 7) * TI;   // image-group base
    const int i0  = (blockIdx.x & 127) * TC;  // caption-group base

    for (int p = tid; p < TI * RR * RR; p += NTH)
        Gs[p] = G[(size_t)b0 * RR * RR + p];
    for (int p = tid; p < TC * LLW * LLW; p += NTH)
        Hs[p] = H[(size_t)i0 * LLW * LLW + p];

    const int wave = tid >> 5;
    const int lane = tid & 31;
    const int grp  = lane >> 4;
    const int lr   = lane & 15;
    const int m0   = (wave / 3) * 16;
    const int n0   = (wave % 3) * 16;

    v8f acc[TI][TC] = {};

    for (int c = 0; c < DD / KC; ++c) {
        __syncthreads();
        // ---- stage A (2 images x 36 rows x 64 halfs) as 16B units ----
        for (int u = tid; u < TI * RR * (KC / 8); u += NTH) {
            int e   = u / (RR * (KC / 8));
            int rem = u - e * (RR * (KC / 8));
            int r   = rem >> 3, kv = rem & 7;
            const _Float16* src =
                imgH + (((size_t)(b0 + e) * RR + r) << 10) + (c * KC) + (kv * 8);
            _Float16* dst = &Ah[(e * 48 + r) * KS + kv * 8];
#ifdef ASYNC_CP
            ASYNC_CP_B128(src, dst);
#else
            *(v8h*)dst = *(const v8h*)src;
#endif
        }
        // ---- stage B (2 captions x 40 rows x 64 halfs) ----
        for (int u = tid; u < TC * LLW * (KC / 8); u += NTH) {
            int f   = u / (LLW * (KC / 8));
            int rem = u - f * (LLW * (KC / 8));
            int l   = rem >> 3, kv = rem & 7;
            const _Float16* src =
                capH + (((size_t)(i0 + f) * LLW + l) << 10) + (c * KC) + (kv * 8);
            _Float16* dst = &Bh[(f * 48 + l) * KS + kv * 8];
#ifdef ASYNC_CP
            ASYNC_CP_B128(src, dst);
#else
            *(v8h*)dst = *(const v8h*)src;
#endif
        }
#ifdef ASYNC_CP
#if __has_builtin(__builtin_amdgcn_s_wait_asynccnt)
        __builtin_amdgcn_s_wait_asynccnt(0);
#else
        asm volatile("s_wait_asynccnt 0x0" ::: "memory");
#endif
#endif
        __syncthreads();
        if (wave < 9) {
            #pragma unroll
            for (int kk = 0; kk < KC; kk += 32) {
                // A frag (ISA 7.12.2): elems 0-7 @ K=kk+8*grp, elems 8-15 @ +16.
                v16h a[TI], bf[TC];
                #pragma unroll
                for (int e = 0; e < TI; ++e) {
                    const _Float16* ap = &Ah[(e * 48 + m0 + lr) * KS + kk + grp * 8];
                    ((v8h*)&a[e])[0] = *(const v8h*)ap;
                    ((v8h*)&a[e])[1] = *(const v8h*)(ap + 16);
                }
                // B frag: col = lane&15; elems 0-7 @ K=kk+8*grp, elems 8-15 @ +16.
                #pragma unroll
                for (int f = 0; f < TC; ++f) {
                    const _Float16* bp = &Bh[(f * 48 + n0 + lr) * KS + kk + grp * 8];
                    ((v8h*)&bf[f])[0] = *(const v8h*)bp;
                    ((v8h*)&bf[f])[1] = *(const v8h*)(bp + 16);
                }
                #pragma unroll
                for (int e = 0; e < TI; ++e)
                    #pragma unroll
                    for (int f = 0; f < TC; ++f)
                        acc[e][f] = __builtin_amdgcn_wmma_f32_16x16x32_f16(
                            false, a[e], false, bf[f], (short)0, acc[e][f],
                            false, false);
            }
        }
    }
    __syncthreads();
    if (wave < 9) {
        // C layout: VGPR j -> row m0 + j + 8*grp, col n0 + (lane&15).
        #pragma unroll
        for (int e = 0; e < TI; ++e)
            #pragma unroll
            for (int f = 0; f < TC; ++f)
                #pragma unroll
                for (int j = 0; j < 8; ++j)
                    Sl[((e * TC + f) * 48 + m0 + grp * 8 + j) * SS + n0 + lr] =
                        acc[e][f][j];
    }
    __syncthreads();

    // ---- postprocessing: 80 threads per (image,caption) pair ----
    const int p  = tid / 80;          // pair id: e = p>>1 image, f = p&1 caption
    const int t  = tid - p * 80;
    const int e  = p >> 1, f = p & 1;
    const int bb = b0 + e, ii = i0 + f;
    const float* Sp = &Sl[p * 48 * SS];
    const float* Gp = &Gs[e * RR * RR];
    const float* Hp = &Hs[f * LLW * LLW];

    if (t < RR) {                       // row norms over words (axis=L)
        float s = 0.0f;
        #pragma unroll
        for (int l = 0; l < LLW; ++l) {
            float v = Sp[t * SS + l]; v = v > 0.0f ? v : 0.1f * v;
            s = fmaf(v, v, s);
        }
        invRow[p * RR + t] = 1.0f / (sqrtf(s) + 1e-8f);
    }
    if (t >= RR && t < RR + LLW) {      // col norms over regions (axis=R)
        int l = t - RR;
        float s = 0.0f;
        #pragma unroll
        for (int r = 0; r < RR; ++r) {
            float v = Sp[r * SS + l]; v = v > 0.0f ? v : 0.1f * v;
            s = fmaf(v, v, s);
        }
        invCol[p * LLW + l] = 1.0f / (sqrtf(s) + 1e-8f);
    }
    __syncthreads();

    if (t < LLW) {                      // ---- t2i: one word column each ----
        const int l = t;
        float a[RR];
        float mx = -3.4e38f;
        #pragma unroll
        for (int r = 0; r < RR; ++r) {
            float v = Sp[r * SS + l]; v = v > 0.0f ? v : 0.1f * v;
            float z = 9.0f * v * invRow[p * RR + r];
            a[r] = z; mx = fmaxf(mx, z);
        }
        float sum = 0.0f;
        #pragma unroll
        for (int r = 0; r < RR; ++r) { a[r] = __expf(a[r] - mx); sum += a[r]; }
        const float inv = 1.0f / sum;
        float w12 = 0.0f, w2sq = 0.0f;
        #pragma unroll
        for (int r = 0; r < RR; ++r) {
            w12 = fmaf(a[r], P[((size_t)bb * RR + r) * NB + ii], w12);
            float tq = 0.0f;
            #pragma unroll
            for (int r2 = 0; r2 < RR; ++r2)
                tq = fmaf(Gp[r * RR + r2], a[r2], tq);
            w2sq = fmaf(a[r], tq, w2sq);
        }
        w12 *= inv; w2sq *= inv * inv;   // ||wctx_l||^2 = attn^T G attn
        redT[p * LLW + l] =
            w12 / fmaxf(w1txt[ii] * sqrtf(w2sq), 1e-8f);
    }
    if (t >= LLW && t < LLW + RR) {     // ---- i2t: one region row each ----
        const int r = t - LLW;
        float a[LLW];
        float mx = -3.4e38f;
        #pragma unroll
        for (int l = 0; l < LLW; ++l) {
            float v = Sp[r * SS + l]; v = v > 0.0f ? v : 0.1f * v;
            float z = 9.0f * v * invCol[p * LLW + l];
            a[l] = z; mx = fmaxf(mx, z);
        }
        float sum = 0.0f;
        #pragma unroll
        for (int l = 0; l < LLW; ++l) { a[l] = __expf(a[l] - mx); sum += a[l]; }
        const float inv = 1.0f / sum;
        float w12 = 0.0f, w2sq = 0.0f;
        #pragma unroll
        for (int l = 0; l < LLW; ++l) {
            w12 = fmaf(a[l], Q[((size_t)ii * LLW + l) * NB + bb], w12);
            float tq = 0.0f;
            #pragma unroll
            for (int l2 = 0; l2 < LLW; ++l2)
                tq = fmaf(Hp[l * LLW + l2], a[l2], tq);
            w2sq = fmaf(a[l], tq, w2sq);
        }
        w12 *= inv; w2sq *= inv * inv;
        redI[p * RR + r] =
            w12 / fmaxf(w1img[bb] * sqrtf(w2sq), 1e-8f);
    }
    __syncthreads();

    if (t == 0) {                       // one thread per pair writes output
        float mt = 0.0f;
        for (int l = 0; l < LLW; ++l) mt += redT[p * LLW + l];
        mt *= (1.0f / LLW);
        float mi = 0.0f;
        for (int r = 0; r < RR; ++r) mi += redI[p * RR + r];
        mi *= (1.0f / RR);
        float t2i = logf(expf(mt * 6.0f)) * (1.0f / 6.0f);  // literal LSE
        float i2t = logf(expf(mi * 6.0f)) * (1.0f / 6.0f);
        out[(size_t)bb * NB + ii] = t2i + i2t + basem[(size_t)bb * NB + ii];
    }
}

// ---------------------------------------------------------------------------
extern "C" void kernel_launch(void* const* d_in, const int* in_sizes, int n_in,
                              void* d_out, int out_size, void* d_ws, size_t ws_size,
                              hipStream_t stream)
{
    (void)in_sizes; (void)n_in; (void)out_size; (void)ws_size;
    const float* pool_img = (const float*)d_in[0];   // (256,1024)
    const float* img_emb  = (const float*)d_in[1];   // (256,36,1024)
    const float* pool_txt = (const float*)d_in[2];   // (256,1024)
    const float* cap_emb  = (const float*)d_in[3];   // (256,40,1024)
    float* out = (float*)d_out;

    // Workspace layout: fp32 tables (~22.2 MB) then f16 copies (~40 MB).
    float* P     = (float*)d_ws;                       // 9216*256
    float* Q     = P + (size_t)NB * RR * NB;           // 10240*256
    float* G     = Q + (size_t)NB * LLW * NB;          // 256*36*36
    float* H     = G + (size_t)NB * RR * RR;           // 256*40*40
    float* basem = H + (size_t)NB * LLW * LLW;         // 256*256
    float* w1i   = basem + (size_t)NB * NB;            // 256
    float* w1t   = w1i + NB;                           // 256
    _Float16* imgH = (_Float16*)(w1t + NB);            // 9216*1024 f16
    _Float16* capH = imgH + (size_t)NB * RR * DD;      // 10240*1024 f16

    cvt_f16_kernel<<<2048, 256, 0, stream>>>(img_emb, imgH, NB * RR * DD);
    cvt_f16_kernel<<<2048, 256, 0, stream>>>(cap_emb, capH, NB * LLW * DD);
    norm_kernel<<<2, 256, 0, stream>>>(pool_img, pool_txt, w1i, w1t);
    dot_rv_kernel<<<NB, 256, 0, stream>>>(pool_img, pool_txt, basem);
    dot_rv_kernel<<<NB * RR, 256, 0, stream>>>(img_emb, pool_txt, P);
    dot_rv_kernel<<<NB * LLW, 256, 0, stream>>>(cap_emb, pool_img, Q);
    gram_kernel<<<NB, 256, 0, stream>>>(img_emb, G, RR);
    gram_kernel<<<NB, 256, 0, stream>>>(cap_emb, H, LLW);

    pair_kernel<<<(NB / TI) * (NB / TC), NTH, 0, stream>>>(
        imgH, capH, P, Q, G, H, basem, w1i, w1t, out);
}
